// KOSS_34265249087558
// MI455X (gfx1250) — matrix operand order repaired and checked by reading, not compiled, not run on hardware
//
#include <hip/hip_runtime.h>
#include <math.h>

typedef _Float16 f16;
typedef __attribute__((ext_vector_type(16))) _Float16 v16h;
typedef __attribute__((ext_vector_type(8)))  _Float16 v8h;
typedef __attribute__((ext_vector_type(8)))  float    v8f;
typedef unsigned int v4u __attribute__((ext_vector_type(4)));
typedef int          v8i __attribute__((ext_vector_type(8)));
typedef int          v4i __attribute__((ext_vector_type(4)));

#if __has_builtin(__builtin_amdgcn_tensor_load_to_lds)
#define KOSS_HAS_TDM 1
#else
#define KOSS_HAS_TDM 0
#endif

namespace {
constexpr int Bc    = 4;
constexpr int Lc    = 2048;
constexpr int DMc   = 1024;
constexpr int EDc   = 2048;
constexpr int NSc   = 2;      // state dim N
constexpr int DTRc  = 64;
constexpr int SEGc  = 256;
constexpr int NSEGc = 8;
constexpr int XZLD  = 2 * EDc;   // 4096
constexpr int DCN   = DTRc + NSc; // 66
constexpr float PIF = 3.14159265358979323846f;
constexpr int EPT   = EDc / 256;
}

// ---------------------------------------------------------------------------
// TDM: issue a 2D tile load (rows x 32 f16, row stride = strideElems) into LDS.
// D# built per CDNA5 ISA (normal mode, 2D).
// ---------------------------------------------------------------------------
#if KOSS_HAS_TDM
__device__ __forceinline__ void koss_tdm_load(const f16* gptr, unsigned lds_off,
                                              int rows, int tensorD0, long strideElems)
{
    unsigned long long ga = (unsigned long long)(uintptr_t)gptr;
    unsigned long long st = (unsigned long long)strideElems;
    v4u g0;
    g0[0] = 1u;                                          // count=1, user desc
    g0[1] = lds_off;                                     // lds_addr (bytes)
    g0[2] = (unsigned)(ga & 0xFFFFFFFFu);                // global_addr[31:0]
    g0[3] = (unsigned)((ga >> 32) & 0x01FFFFFFu) | 0x80000000u; // addr[56:32] | type=2
    v8i g1;
    g1[0] = 0x00010000;                                  // wg_mask=0, data_size=2B
    g1[1] = (int)(((unsigned)tensorD0 & 0xFFFFu) << 16); // tensor_dim0[15:0]
    g1[2] = (int)((((unsigned)tensorD0 >> 16) & 0xFFFFu) | 0x7FFF0000u); // dim0 hi | dim1 lo (big)
    g1[3] = (int)(32u << 16);                            // tensor_dim1 hi=0 | tile_dim0=32
    g1[4] = rows & 0xFFFF;                               // tile_dim1=rows, tile_dim2=0
    g1[5] = (int)(st & 0xFFFFFFFFu);                     // stride0[31:0] (elems)
    g1[6] = (int)((st >> 32) & 0xFFFFu);                 // stride0[47:32] | stride1 lo=0
    g1[7] = 0;
    v4i z4; z4[0] = 0; z4[1] = 0; z4[2] = 0; z4[3] = 0;  // groups 2/3 unused (2D)
#if defined(__clang_major__) && __clang_major__ >= 23
    v8i z8;
    #pragma unroll
    for (int i = 0; i < 8; ++i) z8[i] = 0;
    __builtin_amdgcn_tensor_load_to_lds(g0, g1, z4, z4, z8, 0);
#else
    __builtin_amdgcn_tensor_load_to_lds(g0, g1, z4, z4, 0);
#endif
}
#endif

// ---------------------------------------------------------------------------
// Generic WMMA GEMM:  out = act( A[M,K](f16) @ W[N,K]^T(f16) + bias )
// Block: 256 threads = 8 waves. Tile 128(M) x 64(N), K-step 32.
// Wave w computes rows [w*16, w*16+16) x all 64 cols (4x v_wmma_f32_16x16x32_f16).
// Full tiles: double-buffered TDM (tensor_load_to_lds + s_wait_tensorcnt).
// Partial tiles (N=66 GEMM edge): guarded manual staging.
// ACT: 0=none, 1=relu, 2=softplus
// ---------------------------------------------------------------------------
template<int ACT, bool HAS_BIAS, bool RESIDUAL>
__global__ void __launch_bounds__(256)
gemm_wmma(const f16* __restrict__ A, const f16* __restrict__ W,
          const float* __restrict__ bias,
          float* __restrict__ out32, f16* __restrict__ out16,
          int M, int N, int K)
{
    constexpr int TM = 128, TN = 64, TK = 32;
    __shared__ __align__(32) f16 sA[2][TM * TK];
    __shared__ __align__(32) f16 sB[2][TN * TK];

    const int tid  = threadIdx.x;
    const int lane = tid & 31;
    const int wave = tid >> 5;
    const int mBase = blockIdx.y * TM;
    const int nBase = blockIdx.x * TN;

    v8f acc[4];
    #pragma unroll
    for (int i = 0; i < 4; ++i)
        #pragma unroll
        for (int j = 0; j < 8; ++j) acc[i][j] = 0.0f;

    // one K-step of WMMA out of LDS buffer `buf` (ISA 7.12.2 fragment layouts)
    auto compute = [&](int buf) {
        const int am = lane & 15;
        const int ka = (lane < 16) ? 0 : 16;
        const v8h* pa = (const v8h*)&sA[buf][(wave * 16 + am) * TK + ka];
        v8h alo = pa[0];
        v8h ahi = pa[1];   // +8 halves
        v16h afrag;
        #pragma unroll
        for (int i = 0; i < 8; ++i) { afrag[i] = alo[i]; afrag[i + 8] = ahi[i]; }
        const int kb = (lane < 16) ? 0 : 16;
        #pragma unroll
        for (int nt = 0; nt < 4; ++nt) {
            v16h bfrag = *(const v16h*)&sB[buf][(nt * 16 + am) * TK + kb];
            acc[nt] = __builtin_amdgcn_wmma_f32_16x16x32_f16(
                false, afrag, false, bfrag, (short)0, acc[nt], false, false);
        }
    };

#if KOSS_HAS_TDM
    const bool full = (mBase + TM <= M) && (nBase + TN <= N);
#else
    const bool full = false;
#endif

    if (full) {
#if KOSS_HAS_TDM
        // async DMA pipeline: wave 0 drives the Tensor Data Mover
        if (wave == 0) {
            koss_tdm_load(A + (size_t)mBase * K, (unsigned)(uintptr_t)&sA[0][0], TM, K, K);
            koss_tdm_load(W + (size_t)nBase * K, (unsigned)(uintptr_t)&sB[0][0], TN, K, K);
        }
        int buf = 0;
        for (int k0 = 0; k0 < K; k0 += TK) {
            if (wave == 0) __builtin_amdgcn_s_wait_tensorcnt(0);
            __syncthreads();           // cur buf ready; prev readers of other buf done
            if (wave == 0 && k0 + TK < K) {
                koss_tdm_load(A + (size_t)mBase * K + k0 + TK,
                              (unsigned)(uintptr_t)&sA[buf ^ 1][0], TM, K, K);
                koss_tdm_load(W + (size_t)nBase * K + k0 + TK,
                              (unsigned)(uintptr_t)&sB[buf ^ 1][0], TN, K, K);
            }
            compute(buf);
            buf ^= 1;
        }
#endif
    } else {
        // guarded manual staging (edge tiles only: the N=66 projection)
        for (int k0 = 0; k0 < K; k0 += TK) {
            #pragma unroll
            for (int c = 0; c < 2; ++c) {
                int idx = (tid + c * 256) * 8;   // A tile: 4096 halves
                int r   = idx >> 5;
                int col = idx & 31;
                v8h va;
                #pragma unroll
                for (int i = 0; i < 8; ++i) va[i] = (f16)0;
                int gm = mBase + r;
                if (gm < M) va = *(const v8h*)&A[(size_t)gm * K + k0 + col];
                *(v8h*)&sA[0][idx] = va;
            }
            {
                int idx = tid * 8;               // B tile: 2048 halves
                int r   = idx >> 5;
                int col = idx & 31;
                v8h vb;
                #pragma unroll
                for (int i = 0; i < 8; ++i) vb[i] = (f16)0;
                int gn = nBase + r;
                if (gn < N) vb = *(const v8h*)&W[(size_t)gn * K + k0 + col];
                *(v8h*)&sB[0][idx] = vb;
            }
            if (k0 + TK < K)
                __builtin_prefetch(&A[(size_t)(mBase + (tid >> 2)) * K + k0 + TK], 0, 0);
            __syncthreads();
            compute(0);
            __syncthreads();
        }
    }

    // Epilogue: C/D layout: elem r -> row = r + (lane>=16 ? 8:0), col = lane%16
    const int nloc = lane & 15;
    const int moff = (lane < 16) ? 0 : 8;
    #pragma unroll
    for (int nt = 0; nt < 4; ++nt) {
        int col = nBase + nt * 16 + nloc;
        if (col >= N) continue;
        float bv = HAS_BIAS ? bias[col] : 0.0f;
        #pragma unroll
        for (int r = 0; r < 8; ++r) {
            int row = mBase + wave * 16 + moff + r;
            if (row >= M) continue;
            float v = acc[nt][r] + bv;
            if (ACT == 1)       v = fmaxf(v, 0.0f);
            else if (ACT == 2)  v = (v > 20.0f) ? v : log1pf(__expf(v));
            size_t o = (size_t)row * N + col;
            if (RESIDUAL) { if (out32) out32[o] += v; }
            else          { if (out32) out32[o]  = v; }
            if (out16) out16[o] = (f16)v;
        }
    }
}

// ---------------------------------------------------------------------------
// Elementwise / row kernels
// ---------------------------------------------------------------------------
__global__ void koss_cast16(const float* __restrict__ s, f16* __restrict__ d, long n)
{
    long i = (long)blockIdx.x * blockDim.x + threadIdx.x;
    if (i < n) d[i] = (f16)s[i];
}

__global__ void koss_fill0(float* __restrict__ p, long n)
{
    long i = (long)blockIdx.x * blockDim.x + threadIdx.x;
    if (i < n) p[i] = 0.0f;
}

__global__ void koss_rmsnorm(const float* __restrict__ h, const float* __restrict__ w,
                             f16* __restrict__ xn)
{
    __shared__ float red[256];
    const int row = blockIdx.x;           // B*L rows
    const int tid = threadIdx.x;
    float s = 0.0f;
    #pragma unroll
    for (int j = 0; j < DMc / 256; ++j) {
        float v = h[(size_t)row * DMc + tid + j * 256];
        s += v * v;
    }
    red[tid] = s;
    __syncthreads();
    for (int sh = 128; sh > 0; sh >>= 1) {
        if (tid < sh) red[tid] += red[tid + sh];
        __syncthreads();
    }
    float rs = rsqrtf(red[0] / DMc + 1e-5f);
    #pragma unroll
    for (int j = 0; j < DMc / 256; ++j) {
        int c = tid + j * 256;
        xn[(size_t)row * DMc + c] = (f16)(h[(size_t)row * DMc + c] * rs * w[c]);
    }
}

// fused: two depthwise convs (DCONV=2, causal pad) + SiLU, from xz[:, :ED]
__global__ void koss_dwconv_silu(const float* __restrict__ xz,
                                 const float* __restrict__ c1w, const float* __restrict__ c1b,
                                 const float* __restrict__ c2w, const float* __restrict__ c2b,
                                 float* __restrict__ xs32, f16* __restrict__ xs16)
{
    long idx = (long)blockIdx.x * blockDim.x + threadIdx.x;
    if (idx >= (long)Bc * Lc * EDc) return;
    int  e   = (int)(idx % EDc);
    long row = idx / EDc;                 // b*L + l
    int  l   = (int)(row % Lc);
    float x0  = xz[row * XZLD + e];
    float xm1 = (l >= 1) ? xz[(row - 1) * XZLD + e] : 0.0f;
    float xm2 = (l >= 2) ? xz[(row - 2) * XZLD + e] : 0.0f;
    float w10 = c1w[e * 2], w11 = c1w[e * 2 + 1], b1 = c1b[e];
    float w20 = c2w[e * 2], w21 = c2w[e * 2 + 1], b2 = c2b[e];
    float y1  = b1 + w10 * xm1 + w11 * x0;
    float y1m = (l >= 1) ? (b1 + w10 * xm2 + w11 * xm1) : 0.0f;  // conv1 out is zero-padded
    float y2  = b2 + w20 * y1m + w21 * y1;
    float sv  = y2 / (1.0f + __expf(-y2));
    xs32[row * EDc + e] = sv;
    xs16[row * EDc + e] = (f16)sv;
}

__global__ void koss_split_dr(const float* __restrict__ dC, f16* __restrict__ dr16)
{
    long idx = (long)blockIdx.x * blockDim.x + threadIdx.x;
    if (idx >= (long)Bc * Lc * DTRc) return;
    long row = idx / DTRc;
    int  c   = (int)(idx % DTRc);
    dr16[idx] = (f16)dC[row * DCN + c];
}

// Kerr = softsign((xg - yh)^2), f16 for the kproj GEMM chain
__global__ void koss_kerr(const float* __restrict__ xs, const float* __restrict__ yh,
                          int yh_l0, int yh_ldL, int s0, f16* __restrict__ e16)
{
    long idx = (long)blockIdx.x * blockDim.x + threadIdx.x;
    if (idx >= (long)Bc * SEGc * EDc) return;
    int  e = (int)(idx % EDc);
    long r = idx / EDc;                   // b*SEG + t
    int  t = (int)(r % SEGc);
    int  b = (int)(r / SEGc);
    float xg = xs[((size_t)b * Lc + s0 + t) * EDc + e];
    float yv = yh[((size_t)b * yh_ldL + yh_l0 + t) * EDc + e];
    float u  = xg - yv; u = u * u;
    e16[r * EDc + e] = (f16)(u / (1.0f + u));
}

// layernorm of kproj output + EMA update of K with clipped alpha
__global__ void koss_lnK(const float* __restrict__ h3, const float* __restrict__ g,
                         const float* __restrict__ bb, const float* __restrict__ k_alpha,
                         float* __restrict__ Kbuf)
{
    __shared__ float r1[256];
    __shared__ float r2[256];
    const int row = blockIdx.x;           // B*SEG rows
    const int tid = threadIdx.x;
    float s = 0.0f, s2 = 0.0f;
    #pragma unroll
    for (int j = 0; j < EPT; ++j) {
        float v = h3[(size_t)row * EDc + tid + j * 256];
        s += v; s2 += v * v;
    }
    r1[tid] = s; r2[tid] = s2;
    __syncthreads();
    for (int sh = 128; sh > 0; sh >>= 1) {
        if (tid < sh) { r1[tid] += r1[tid + sh]; r2[tid] += r2[tid + sh]; }
        __syncthreads();
    }
    float mu  = r1[0] / EDc;
    float var = r2[0] / EDc - mu * mu;
    float rs  = rsqrtf(var + 1e-5f);
    #pragma unroll
    for (int j = 0; j < EPT; ++j) {
        int e = tid + j * 256;
        float kn = g[e] * (h3[(size_t)row * EDc + e] - mu) * rs + bb[e];
        float a  = fminf(fmaxf(k_alpha[e], 0.01f), 0.99f);
        size_t o = (size_t)row * EDc + e;
        Kbuf[o] = (1.0f - a) * Kbuf[o] + a * kn;
    }
}

// forward DFT of the segment + freq-domain scaling: Z = i * (2*pi*f*g/(dt+eps)) * Xf
__global__ void koss_fdu1(const float* __restrict__ xs, const float* __restrict__ delta,
                          const float* __restrict__ sigma_p, int s0,
                          float* __restrict__ Zre, float* __restrict__ Zim)
{
    __shared__ float cw[SEGc];
    __shared__ float sw[SEGc];
    const int blk = blockIdx.x;           // b*SEG + k
    const int b   = blk / SEGc;
    const int k   = blk % SEGc;
    const int tid = threadIdx.x;          // 256
    {
        float th = (2.0f * PIF * k / (float)SEGc) * tid;
        cw[tid] = cosf(th); sw[tid] = sinf(th);
    }
    __syncthreads();
    float re[EPT], im[EPT];
    #pragma unroll
    for (int j = 0; j < EPT; ++j) { re[j] = 0.0f; im[j] = 0.0f; }
    for (int t = 0; t < SEGc; ++t) {
        float c = cw[t], s = sw[t];
        size_t row = ((size_t)b * Lc + s0 + t) * EDc;
        #pragma unroll
        for (int j = 0; j < EPT; ++j) {
            float x = xs[row + tid + j * 256];
            re[j] += x * c; im[j] -= x * s;
        }
    }
    float fk   = (k < SEGc / 2) ? (float)k / SEGc : (float)(k - SEGc) / SEGc;
    float sg   = sigma_p[0];
    float gk   = __expf(-fk * fk * sg * sg);
    float base = 2.0f * PIF * fk * gk;
    size_t drow = ((size_t)b * Lc + s0 + k) * EDc;
    size_t srow = ((size_t)b * SEGc + k) * EDc;
    #pragma unroll
    for (int j = 0; j < EPT; ++j) {
        int e = tid + j * 256;
        float cc = base / (delta[drow + e] + 1e-5f);
        Zre[srow + e] = -cc * im[j];
        Zim[srow + e] =  cc * re[j];
    }
}

// inverse DFT (real part) * K -> Kdy
__global__ void koss_fdu2(const float* __restrict__ Zre, const float* __restrict__ Zim,
                          const float* __restrict__ Kbuf, float* __restrict__ Kdy)
{
    __shared__ float cw[SEGc];
    __shared__ float sw[SEGc];
    const int blk = blockIdx.x;           // b*SEG + t
    const int b   = blk / SEGc;
    const int t   = blk % SEGc;
    const int tid = threadIdx.x;
    {
        float th = (2.0f * PIF * t / (float)SEGc) * tid;
        cw[tid] = cosf(th); sw[tid] = sinf(th);
    }
    __syncthreads();
    float acc[EPT];
    #pragma unroll
    for (int j = 0; j < EPT; ++j) acc[j] = 0.0f;
    for (int k = 0; k < SEGc; ++k) {
        float c = cw[k], s = sw[k];
        size_t srow = ((size_t)b * SEGc + k) * EDc;
        #pragma unroll
        for (int j = 0; j < EPT; ++j) {
            int e = tid + j * 256;
            acc[j] += Zre[srow + e] * c - Zim[srow + e] * s;
        }
    }
    size_t orow = ((size_t)b * SEGc + t) * EDc;
    const float inv = 1.0f / (float)SEGc;
    #pragma unroll
    for (int j = 0; j < EPT; ++j) {
        int e = tid + j * 256;
        Kdy[orow + e] = Kbuf[orow + e] * acc[j] * inv;
    }
}

// selective-scan over one segment (2 states), writes yh directly into yfull
__global__ void koss_scan(const float* __restrict__ Kbuf, const float* __restrict__ dC,
                          const float* __restrict__ xs, const float* __restrict__ delta,
                          const float* __restrict__ Kdy, const float* __restrict__ A_log,
                          const float* __restrict__ Dp, float* __restrict__ yfull, int s0)
{
    int tid = blockIdx.x * blockDim.x + threadIdx.x;
    if (tid >= Bc * EDc) return;
    const int b = tid / EDc;
    const int e = tid % EDc;
    const float A0 = -__expf(A_log[e * NSc + 0]);
    const float A1 = -__expf(A_log[e * NSc + 1]);
    const float dp = Dp[e];
    float h0 = 0.0f, h1 = 0.0f;           // scan restarts each segment (ref semantics)
    for (int t = 0; t < SEGc; ++t) {
        size_t row  = (size_t)b * Lc + s0 + t;
        size_t srow = (size_t)b * SEGc + t;
        float Kv  = Kbuf[srow * EDc + e];
        float C0  = dC[row * DCN + DTRc + 0];
        float C1  = dC[row * DCN + DTRc + 1];
        float xg  = xs[row * EDc + e];
        float dg  = delta[row * EDc + e];
        float kdy = Kdy[srow * EDc + e];
        float KC0 = Kv * C0, KC1 = Kv * C1;
        float m0 = A0 * (1.0f - KC0), m1 = A1 * (1.0f - KC1);
        float dA0 = __expf(dg * m0 * (1.0f + KC0));
        float dA1 = __expf(dg * m1 * (1.0f + KC1));
        h0 = dA0 * h0 + (dg * (-m0 * Kv) * xg + kdy);
        h1 = dA1 * h1 + (dg * (-m1 * Kv) * xg + kdy);
        yfull[row * EDc + e] = h0 * C0 + h1 * C1 + dp * xg;
    }
}

// gate = y * silu(z)  (z = xz[:, ED:2ED]) -> f16 input for the output projection
__global__ void koss_gate(const float* __restrict__ yfull, const float* __restrict__ xz,
                          f16* __restrict__ g16)
{
    long idx = (long)blockIdx.x * blockDim.x + threadIdx.x;
    if (idx >= (long)Bc * Lc * EDc) return;
    int  e   = (int)(idx % EDc);
    long row = idx / EDc;
    float z  = xz[row * XZLD + EDc + e];
    float sv = z / (1.0f + __expf(-z));
    g16[row * EDc + e] = (f16)(yfull[row * EDc + e] * sv);
}

// ---------------------------------------------------------------------------
extern "C" void kernel_launch(void* const* d_in, const int* in_sizes, int n_in,
                              void* d_out, int out_size, void* d_ws, size_t ws_size,
                              hipStream_t stream)
{
    (void)in_sizes; (void)n_in; (void)out_size; (void)ws_size;
    const float* x_in  = (const float*)d_in[0];
    const float* rms_w = (const float*)d_in[1];
    const float* in_w  = (const float*)d_in[2];
    const float* c1_w  = (const float*)d_in[3];
    const float* c1_b  = (const float*)d_in[4];
    const float* c2_w  = (const float*)d_in[5];
    const float* c2_b  = (const float*)d_in[6];
    const float* xp_w  = (const float*)d_in[7];
    const float* dt_w  = (const float*)d_in[8];
    const float* dt_b  = (const float*)d_in[9];
    const float* A_log = (const float*)d_in[10];
    const float* Dp    = (const float*)d_in[11];
    const float* out_w = (const float*)d_in[12];
    const float* k1_w  = (const float*)d_in[13];
    const float* k1_b  = (const float*)d_in[14];
    const float* k2_w  = (const float*)d_in[15];
    const float* k3_w  = (const float*)d_in[16];
    const float* ln_g  = (const float*)d_in[17];
    const float* ln_b  = (const float*)d_in[18];
    const float* k_al  = (const float*)d_in[19];
    const float* sigma = (const float*)d_in[20];
    float* h = (float*)d_out;             // residual stream lives in d_out

    const long ROWS = (long)Bc * Lc;      // 8192
    const long SROW = (long)Bc * SEGc;    // 1024

    char* base = (char*)d_ws;
    size_t off = 0;
    auto alloc = [&](size_t bytes) -> void* {
        off = (off + 255) & ~(size_t)255;
        void* p = base + off; off += bytes; return p;
    };
    f16*   xn16   = (f16*)  alloc(ROWS * DMc * 2);
    float* xz32   = (float*)alloc(ROWS * XZLD * 4);
    float* xs32   = (float*)alloc(ROWS * EDc * 4);
    f16*   xs16   = (f16*)  alloc(ROWS * EDc * 2);
    float* dC32   = (float*)alloc(ROWS * DCN * 4);
    f16*   dr16   = (f16*)  alloc(ROWS * DTRc * 2);
    float* del32  = (float*)alloc(ROWS * EDc * 4);
    float* Kbuf   = (float*)alloc(SROW * EDc * 4);
    f16*   e16    = (f16*)  alloc(SROW * EDc * 2);
    f16*   h1_16  = (f16*)  alloc(SROW * (size_t)3 * EDc * 2);
    f16*   h2_16  = (f16*)  alloc(SROW * EDc * 2);
    float* h3_32  = (float*)alloc(SROW * EDc * 4);
    float* Zre    = (float*)alloc(SROW * EDc * 4);
    float* Zim    = (float*)alloc(SROW * EDc * 4);
    float* Kdy    = (float*)alloc(SROW * EDc * 4);
    float* yfull  = (float*)alloc(ROWS * EDc * 4);
    f16*   g16    = (f16*)  alloc(ROWS * EDc * 2);
    float* yh0    = (float*)alloc(SROW * EDc * 4);
    f16*   in_w16 = (f16*)  alloc((size_t)2 * EDc * DMc * 2);
    f16*   xp_w16 = (f16*)  alloc((size_t)DCN * EDc * 2);
    f16*   dt_w16 = (f16*)  alloc((size_t)EDc * DTRc * 2);
    f16*   k1_16  = (f16*)  alloc((size_t)3 * EDc * EDc * 2);
    f16*   k2_16  = (f16*)  alloc((size_t)EDc * 3 * EDc * 2);
    f16*   k3_16  = (f16*)  alloc((size_t)EDc * EDc * 2);
    f16*   ow16   = (f16*)  alloc((size_t)DMc * EDc * 2);

    hipMemcpyAsync(h, x_in, sizeof(float) * ROWS * DMc, hipMemcpyDeviceToDevice, stream);

    auto gblk = [](long n) { return (unsigned)((n + 255) / 256); };
    koss_fill0<<<gblk(SROW * EDc), 256, 0, stream>>>(yh0, SROW * EDc);

    for (int i = 0; i < 2; ++i) {
        // weight conversion f32 -> f16 (small vs GEMM cost)
        long nw;
        nw = (long)2 * EDc * DMc;
        koss_cast16<<<gblk(nw), 256, 0, stream>>>(in_w + (size_t)i * nw, in_w16, nw);
        nw = (long)DCN * EDc;
        koss_cast16<<<gblk(nw), 256, 0, stream>>>(xp_w + (size_t)i * nw, xp_w16, nw);
        nw = (long)EDc * DTRc;
        koss_cast16<<<gblk(nw), 256, 0, stream>>>(dt_w + (size_t)i * nw, dt_w16, nw);
        nw = (long)3 * EDc * EDc;
        koss_cast16<<<gblk(nw), 256, 0, stream>>>(k1_w + (size_t)i * nw, k1_16, nw);
        koss_cast16<<<gblk(nw), 256, 0, stream>>>(k2_w + (size_t)i * nw, k2_16, nw);
        nw = (long)EDc * EDc;
        koss_cast16<<<gblk(nw), 256, 0, stream>>>(k3_w + (size_t)i * nw, k3_16, nw);
        nw = (long)DMc * EDc;
        koss_cast16<<<gblk(nw), 256, 0, stream>>>(out_w + (size_t)i * nw, ow16, nw);

        // RMSNorm -> xn (f16)
        koss_rmsnorm<<<(unsigned)ROWS, 256, 0, stream>>>(h, rms_w + (size_t)i * DMc, xn16);

        // xz = xn @ in_w^T   [8192,1024] x [4096,1024]^T
        {
            dim3 g(XZLD / 64, (unsigned)(ROWS / 128));
            gemm_wmma<0, false, false><<<g, 256, 0, stream>>>(
                xn16, in_w16, nullptr, xz32, nullptr, (int)ROWS, XZLD, DMc);
        }
        // double depthwise conv + SiLU
        koss_dwconv_silu<<<gblk(ROWS * EDc), 256, 0, stream>>>(
            xz32, c1_w + (size_t)i * EDc * 2, c1_b + (size_t)i * EDc,
            c2_w + (size_t)i * EDc * 2, c2_b + (size_t)i * EDc, xs32, xs16);

        // dC = xs @ xp_w^T   [8192,2048] x [66,2048]^T
        {
            dim3 g((DCN + 63) / 64, (unsigned)(ROWS / 128));
            gemm_wmma<0, false, false><<<g, 256, 0, stream>>>(
                xs16, xp_w16, nullptr, dC32, nullptr, (int)ROWS, DCN, EDc);
        }
        koss_split_dr<<<gblk(ROWS * DTRc), 256, 0, stream>>>(dC32, dr16);

        // delta = softplus(dr @ dt_w^T + dt_b)   [8192,64] x [2048,64]^T
        {
            dim3 g(EDc / 64, (unsigned)(ROWS / 128));
            gemm_wmma<2, true, false><<<g, 256, 0, stream>>>(
                dr16, dt_w16, dt_b + (size_t)i * EDc, del32, nullptr, (int)ROWS, EDc, DTRc);
        }

        koss_fill0<<<gblk(SROW * EDc), 256, 0, stream>>>(Kbuf, SROW * EDc);

        for (int s = 0; s < NSEGc; ++s) {
            const int s0 = s * SEGc;
            const float* yhp = (s == 0) ? yh0 : yfull;
            const int yl0 = (s == 0) ? 0 : (s0 - SEGc);
            const int yld = (s == 0) ? SEGc : Lc;

            koss_kerr<<<gblk(SROW * EDc), 256, 0, stream>>>(xs32, yhp, yl0, yld, s0, e16);

            // kproj MLP: the FLOP-dominant GEMM chain
            dim3 g1(3 * EDc / 64, (unsigned)(SROW / 128));
            gemm_wmma<1, true, false><<<g1, 256, 0, stream>>>(
                e16, k1_16, k1_b + (size_t)i * 3 * EDc, nullptr, h1_16,
                (int)SROW, 3 * EDc, EDc);
            dim3 g2(EDc / 64, (unsigned)(SROW / 128));
            gemm_wmma<1, false, false><<<g2, 256, 0, stream>>>(
                h1_16, k2_16, nullptr, nullptr, h2_16, (int)SROW, EDc, 3 * EDc);
            gemm_wmma<0, false, false><<<g2, 256, 0, stream>>>(
                h2_16, k3_16, nullptr, h3_32, nullptr, (int)SROW, EDc, EDc);

            koss_lnK<<<(unsigned)SROW, 256, 0, stream>>>(
                h3_32, ln_g + (size_t)i * EDc, ln_b + (size_t)i * EDc,
                k_al + (size_t)i * EDc, Kbuf);

            koss_fdu1<<<(unsigned)SROW, 256, 0, stream>>>(xs32, del32, sigma + i, s0, Zre, Zim);
            koss_fdu2<<<(unsigned)SROW, 256, 0, stream>>>(Zre, Zim, Kbuf, Kdy);

            koss_scan<<<(Bc * EDc + 255) / 256, 256, 0, stream>>>(
                Kbuf, dC32, xs32, del32, Kdy,
                A_log + (size_t)i * EDc * NSc, Dp + (size_t)i * EDc, yfull, s0);
        }

        // out = (y * silu(z)) @ out_w^T, accumulated into h
        koss_gate<<<gblk(ROWS * EDc), 256, 0, stream>>>(yfull, xz32, g16);
        {
            dim3 g(DMc / 64, (unsigned)(ROWS / 128));
            gemm_wmma<0, false, true><<<g, 256, 0, stream>>>(
                g16, ow16, nullptr, h, nullptr, (int)ROWS, DMc, EDc);
        }
    }
}
